// ConditionalCrossAttention_72447508349215
// MI455X (gfx1250) — matrix operand order repaired and checked by reading, not compiled, not run on hardware
//
#include <hip/hip_runtime.h>
#include <hip/hip_bf16.h>

// ---------------------------------------------------------------------------
// Conditional cross-attention (DAB-DETR style) for MI455X / gfx1250.
//
// Roofline: ~30 GFLOP vs ~100 MB mandatory HBM traffic (fp32 K/V/key_pos) ->
// ~6 us memory floor at 23.3 TB/s, comparable compute time on f16 WMMA.
// Strategy:
//   * one-time f32->f16 weight conversion (no v_cvt in GEMM inner loops)
//   * proj_kv blocked 64 rows/block, activations staged once in LDS as f16
//     (HBM reads at roofline minimum; weight L2 traffic amortized 4x),
//     kc/kp/v accumulated in separate K-passes to stay under 256 VGPRs
//   * flash-attention streaming softmax; K/V chunks double-buffered into LDS
//     with gfx1250 async path (global_load_async_to_lds_b128 +
//     s_wait_asynccnt), all matrix math on v_wmma_f32_16x16x32_f16.
// ---------------------------------------------------------------------------

#define DEV __device__ __forceinline__

typedef __attribute__((ext_vector_type(16))) _Float16 v16h;
typedef __attribute__((ext_vector_type(8)))  _Float16 v8h;
typedef __attribute__((ext_vector_type(8)))  float    v8f;

constexpr int BATCH = 8;
constexpr int NQ    = 300;
constexpr int LKV   = 4096;
constexpr int CDIM  = 256;
constexpr int HEADS = 8;
constexpr int DHD   = 32;      // per-head content dim
constexpr int NT    = 19;      // ceil(300/16)
constexpr int NPAD  = NT * 16; // 304
constexpr int WROWS = CDIM * CDIM; // 65536 elems per weight matrix

DEV v8f wmma_f16(v16h a, v16h b, v8f c) {
  return __builtin_amdgcn_wmma_f32_16x16x32_f16(
      false, a, false, b, (short)0, c, false, false);
}

// ---- gfx1250 async global->LDS path ---------------------------------------
DEV void async_ld_b128(void* lds_dst, const void* gsrc) {
  asm volatile("global_load_async_to_lds_b128 %0, %1, off"
               :: "v"((unsigned)(uintptr_t)lds_dst),
                  "v"((unsigned long long)(uintptr_t)gsrc)
               : "memory");
}
template <int N> DEV void wait_async() {
#if __has_builtin(__builtin_amdgcn_s_wait_asynccnt)
  __builtin_amdgcn_s_wait_asynccnt((unsigned short)N);
  asm volatile("" ::: "memory");
#else
  if constexpr (N == 0) asm volatile("s_wait_asynccnt 0x0" ::: "memory");
  else                  asm volatile("s_wait_asynccnt 0xc" ::: "memory");
#endif
}

// A-matrix fragment (16x32 f16, MxK) from an fp32 row (convert on load).
// lanes 0-15: M=lane, K in {ks..ks+15};
// lanes 16-31: M=lane-16, K in {ks+16..ks+31}.
DEV v16h load_a_f32(const float* rowp, int ks, int half) {
  const float* p = rowp + ks + half * 16;
  v16h a;
#pragma unroll
  for (int i = 0; i < 8; ++i) a[i] = (_Float16)p[i];
#pragma unroll
  for (int i = 0; i < 8; ++i) a[8 + i] = (_Float16)p[8 + i];
  return a;
}

// Same A layout from f16 memory (LDS tiles / qcat rows).
DEV v16h load_a_f16(const _Float16* rowp, int kbase, int half) {
  const _Float16* p = rowp + kbase + half * 16;
  v16h a;
#pragma unroll
  for (int i = 0; i < 8; ++i) a[i] = p[i];
#pragma unroll
  for (int i = 0; i < 8; ++i) a[8 + i] = p[8 + i];
  return a;
}

// B-matrix fragment (32x16, KxN) from pre-converted f16 weight W16[n][k]:
// lane gives col n; K contiguous (lanes>=16 start at ks+16).
DEV v16h load_b_w16(const _Float16* W16, int n, int ks, int half) {
  return *(const v16h*)(W16 + (size_t)n * CDIM + ks + half * 16);
}

// 16x16 f32 tile of X(16x256) * W^T, f32 activations + f16 weights.
DEV v8f gemm256(const float* rowp, const _Float16* W16, int ncol, int half) {
  v8f acc = {};
#pragma unroll
  for (int ks = 0; ks < CDIM; ks += 32) {
    v16h a = load_a_f32(rowp, ks, half);
    v16h b = load_b_w16(W16, ncol, ks, half);
    acc = wmma_f16(a, b, acc);
  }
  return acc;
}

DEV float rmax16(float v) {
#pragma unroll
  for (int off = 8; off; off >>= 1) v = fmaxf(v, __shfl_xor(v, off, 32));
  return v;
}
DEV float rsum16(float v) {
#pragma unroll
  for (int off = 8; off; off >>= 1) v += __shfl_xor(v, off, 32);
  return v;
}

// ---------------------------------------------------------------------------
// Kernel 0: one-time weight conversion f32 -> f16 (7 matrices, 458752 elems).
// ---------------------------------------------------------------------------
__global__ void __launch_bounds__(256)
wconv_kernel(const float* __restrict__ Wqc, const float* __restrict__ Wqp,
             const float* __restrict__ Wqs, const float* __restrict__ Wkc,
             const float* __restrict__ Wkp, const float* __restrict__ Wv,
             const float* __restrict__ Wo,  _Float16* __restrict__ W16) {
  const size_t base = ((size_t)blockIdx.x * 256 + threadIdx.x) * 8;
  if (base >= (size_t)7 * WROWS) return;
  const int m = (int)(base >> 16);
  const float* src;
  switch (m) {
    case 0: src = Wqc; break;  case 1: src = Wqp; break;
    case 2: src = Wqs; break;  case 3: src = Wkc; break;
    case 4: src = Wkp; break;  case 5: src = Wv;  break;
    default: src = Wo;
  }
  const float4* p = (const float4*)(src + (base & 65535));
  float4 x0 = p[0], x1 = p[1];
  v8h h;
  h[0] = (_Float16)x0.x; h[1] = (_Float16)x0.y; h[2] = (_Float16)x0.z; h[3] = (_Float16)x0.w;
  h[4] = (_Float16)x1.x; h[5] = (_Float16)x1.y; h[6] = (_Float16)x1.z; h[7] = (_Float16)x1.w;
  *(v8h*)(W16 + base) = h;
}

// ---------------------------------------------------------------------------
// Kernel 1: K/V projections. 64 rows/block (B*L/64 = 512 blocks), 128 thr.
// X (key/key_pos/value) staged once in LDS as f16; each B-fragment reused
// across 4 row tiles. Writes kcat[b][h][l][64] and vT[b][h][d][L] (f16).
// kc/kp/v run as separate K-passes so peak live accumulators stay at 64
// VGPRs (no VGPR-MSB mode, better occupancy than the fused variant).
// ---------------------------------------------------------------------------
DEV v16h lds_a(const _Float16* Xsrc, int rt, int ks, int lane) {
  const int half = lane >> 4;
  return load_a_f16(Xsrc + (size_t)(rt * 16 + (lane & 15)) * CDIM, ks, half);
}

// One 64x16 column-slab: 4 row-tile accumulators over K=256.
DEV void gemm_slab(const _Float16* Xs, const _Float16* W16, int chan, int lane,
                   v8f acc[4]) {
  const int half = lane >> 4;
#pragma unroll
  for (int ks = 0; ks < CDIM; ks += 32) {
    v16h bf = load_b_w16(W16, chan, ks, half);
    v16h a0 = lds_a(Xs, 0, ks, lane);
    v16h a1 = lds_a(Xs, 1, ks, lane);
    v16h a2 = lds_a(Xs, 2, ks, lane);
    v16h a3 = lds_a(Xs, 3, ks, lane);
    acc[0] = wmma_f16(a0, bf, acc[0]);
    acc[1] = wmma_f16(a1, bf, acc[1]);
    acc[2] = wmma_f16(a2, bf, acc[2]);
    acc[3] = wmma_f16(a3, bf, acc[3]);
  }
}

__global__ void __launch_bounds__(128)
proj_kv_kernel(const float* __restrict__ key, const float* __restrict__ key_pos,
               const float* __restrict__ value,
               const _Float16* __restrict__ Wkc16, const _Float16* __restrict__ Wkp16,
               const _Float16* __restrict__ Wv16,
               const float* __restrict__ bkc, const float* __restrict__ bkp,
               const float* __restrict__ bv,
               _Float16* __restrict__ kcat, _Float16* __restrict__ vT) {
  __shared__ _Float16 Xs[3][64 * CDIM];   // 96 KB
  const int tid = threadIdx.x;
  const int grow0 = blockIdx.x * 64;
  const int bb = grow0 >> 12, ll0 = grow0 & (LKV - 1);

  const float* srcs[3] = {key     + (size_t)grow0 * CDIM,
                          key_pos + (size_t)grow0 * CDIM,
                          value   + (size_t)grow0 * CDIM};
#pragma unroll
  for (int s = 0; s < 3; ++s) {
    for (int i = tid; i < 64 * CDIM / 8; i += 128) {
      const float4* p = (const float4*)(srcs[s] + (size_t)i * 8);
      float4 x0 = p[0], x1 = p[1];
      v8h h;
      h[0] = (_Float16)x0.x; h[1] = (_Float16)x0.y; h[2] = (_Float16)x0.z; h[3] = (_Float16)x0.w;
      h[4] = (_Float16)x1.x; h[5] = (_Float16)x1.y; h[6] = (_Float16)x1.z; h[7] = (_Float16)x1.w;
      *(v8h*)(&Xs[s][i * 8]) = h;
    }
  }
  __syncthreads();

  const int lane = tid & 31, wave = tid >> 5, half = lane >> 4;
#pragma unroll 1
  for (int ct = 0; ct < 4; ++ct) {
    const int chan = (wave * 4 + ct) * 16 + (lane & 15);
    const int h = chan >> 5, d = chan & 31;
    _Float16* kbp = kcat + (size_t)(bb * HEADS + h) * LKV * 64;

    // ---- pass 1: kp (kept for the concat + reused in kc+kp) -----------
    v8f akp[4] = {};
    gemm_slab(&Xs[1][0], Wkp16, chan, lane, akp);
    const float bias_kp = bkp[chan];
#pragma unroll
    for (int rt = 0; rt < 4; ++rt)
#pragma unroll
      for (int j = 0; j < 8; ++j) {
        const int l = ll0 + rt * 16 + j + 8 * half;
        kbp[(size_t)l * 64 + 32 + d] = (_Float16)(akp[rt][j] + bias_kp);
      }

    // ---- pass 2: kc, combined with kp ---------------------------------
    v8f akc[4] = {};
    gemm_slab(&Xs[0][0], Wkc16, chan, lane, akc);
    const float bias_k1 = bkc[chan] + bias_kp;
#pragma unroll
    for (int rt = 0; rt < 4; ++rt)
#pragma unroll
      for (int j = 0; j < 8; ++j) {
        const int l = ll0 + rt * 16 + j + 8 * half;
        kbp[(size_t)l * 64 + d] = (_Float16)(akc[rt][j] + akp[rt][j] + bias_k1);
      }

    // ---- pass 3: v, stored transposed vT[b][h][d][L] ------------------
    v8f av[4] = {};
    gemm_slab(&Xs[2][0], Wv16, chan, lane, av);
    const float bias_v = bv[chan];
    _Float16* vbp = vT + ((size_t)(bb * HEADS + h) * DHD + d) * LKV;
#pragma unroll
    for (int rt = 0; rt < 4; ++rt)
#pragma unroll
      for (int j = 0; j < 8; ++j)
        vbp[ll0 + rt * 16 + j + 8 * half] = (_Float16)(av[rt][j] + bias_v);
  }
}

// ---------------------------------------------------------------------------
// Kernel 2: Q projections. grid = B*19, 128 threads.
// qcat[b][h][n][64] = concat(qc+qp, qs) * rsqrt(64), n padded to 304.
// ---------------------------------------------------------------------------
__global__ void __launch_bounds__(128)
proj_q_kernel(const float* __restrict__ query, const float* __restrict__ query_pos,
              const float* __restrict__ query_sine,
              const _Float16* __restrict__ Wqc16, const _Float16* __restrict__ Wqp16,
              const _Float16* __restrict__ Wqs16,
              const float* __restrict__ bqc, const float* __restrict__ bqp,
              const float* __restrict__ bqs, _Float16* __restrict__ qcat) {
  const int lane = threadIdx.x & 31, wave = threadIdx.x >> 5;
  const int half = lane >> 4;
  const int bb = blockIdx.x / NT;
  const int n0 = (blockIdx.x % NT) * 16;
  int srow = n0 + (lane & 15);
  if (srow > NQ - 1) srow = NQ - 1;
  const size_t row = (size_t)bb * NQ + srow;
  const float scale = 0.125f;   // (2*DH)^-0.5

  const float* qp_ = query      + row * CDIM;
  const float* pp_ = query_pos  + row * CDIM;
  const float* sp_ = query_sine + row * CDIM;

  for (int ct = wave; ct < 16; ct += 4) {
    const int chan = ct * 16 + (lane & 15);
    v8f aqc = gemm256(qp_, Wqc16, chan, half);
    v8f aqp = gemm256(pp_, Wqp16, chan, half);
    v8f aqs = gemm256(sp_, Wqs16, chan, half);
    const float b1 = bqc[chan] + bqp[chan];
    const float b2 = bqs[chan];
    const int h = chan >> 5, d = chan & 31;
    _Float16* qb = qcat + (size_t)(bb * HEADS + h) * NPAD * 64;
#pragma unroll
    for (int j = 0; j < 8; ++j) {
      const int n = n0 + j + 8 * half;
      qb[(size_t)n * 64 + d]      = (_Float16)((aqc[j] + aqp[j] + b1) * scale);
      qb[(size_t)n * 64 + 32 + d] = (_Float16)((aqs[j] + b2) * scale);
    }
  }
}

// ---------------------------------------------------------------------------
// Kernel 3: streaming-softmax attention. grid = B*H*19, 128 threads (4 waves).
// Each wave streams 32-wide L chunks (stride 128). K (32x64) and V (32x32)
// chunks are double-buffered into LDS with async global->LDS loads
// (12 x b128 per chunk per wave); s_wait_asynccnt<=12 overlaps the next
// chunk's DMA with the current chunk's WMMA + softmax.
// ---------------------------------------------------------------------------
DEV void stage_chunk(const _Float16* kb, const _Float16* vb, int lc,
                     _Float16* kdst, _Float16* vdst, int lane) {
#pragma unroll
  for (int t = 0; t < 8; ++t) {           // K: 32 rows x 128 B
    const int idx = t * 32 + lane, r = idx >> 3, seg = idx & 7;
    async_ld_b128(kdst + r * 64 + seg * 8,
                  kb + ((size_t)(lc + r)) * 64 + seg * 8);
  }
#pragma unroll
  for (int t = 0; t < 4; ++t) {           // V: 32 d-rows x 64 B
    const int idx = t * 32 + lane, r = idx >> 2, seg = idx & 3;
    async_ld_b128(vdst + r * 32 + seg * 8,
                  vb + (size_t)r * LKV + lc + seg * 8);
  }
}

__global__ void __launch_bounds__(128)
attn_kernel(const _Float16* __restrict__ qcat, const _Float16* __restrict__ kcat,
            const _Float16* __restrict__ vT, float* __restrict__ omid) {
  __shared__ _Float16 Kst[4][2][32 * 64];   // 32 KB
  __shared__ _Float16 Vst[4][2][32 * 32];   // 16 KB
  __shared__ _Float16 Pbuf[4][16 * 32];     //  4 KB
  __shared__ float mbuf[4][16], sbuf[4][16];
  __shared__ float Obuf[4][16][32];         //  8 KB

  const int lane = threadIdx.x & 31, wave = threadIdx.x >> 5;
  const int half = lane >> 4;
  const int nt = blockIdx.x % NT;
  const int bh = blockIdx.x / NT;
  const int n0 = nt * 16;

  const _Float16* qrow = qcat + ((size_t)bh * NPAD + n0 + (lane & 15)) * 64;
  const v16h qa0 = load_a_f16(qrow, 0,  half);
  const v16h qa1 = load_a_f16(qrow, 32, half);

  const _Float16* kb = kcat + (size_t)bh * LKV * 64;
  const _Float16* vb = vT   + (size_t)bh * DHD * LKV;

  float m[8], s[8];
  v8f O0 = {}, O1 = {};
#pragma unroll
  for (int j = 0; j < 8; ++j) { m[j] = -1.0e30f; s[j] = 0.0f; }

  int nbuf = 0;
  stage_chunk(kb, vb, wave * 32, &Kst[wave][0][0], &Vst[wave][0][0], lane);

  for (int lc = wave * 32; lc < LKV; lc += 128) {
    const bool has_next = (lc + 128) < LKV;
    if (has_next) {
      stage_chunk(kb, vb, lc + 128, &Kst[wave][nbuf ^ 1][0], &Vst[wave][nbuf ^ 1][0], lane);
      wait_async<12>();   // previous chunk's 12 transfers retired (FIFO)
    } else {
      wait_async<0>();
    }
    const _Float16* kst = &Kst[wave][nbuf][0];
    const _Float16* vst = &Vst[wave][nbuf][0];

    // ---- S = Q * K^T for 32 keys : 4 WMMAs ----------------------------
    const _Float16* k0 = kst + (lane & 15) * 64 + half * 16;
    const _Float16* k1 = kst + (16 + (lane & 15)) * 64 + half * 16;
    v16h kb00 = *(const v16h*)(k0);
    v16h kb01 = *(const v16h*)(k0 + 32);
    v16h kb10 = *(const v16h*)(k1);
    v16h kb11 = *(const v16h*)(k1 + 32);
    v8f S1 = {}, S2 = {};
    S1 = wmma_f16(qa0, kb00, S1);
    S1 = wmma_f16(qa1, kb01, S1);
    S2 = wmma_f16(qa0, kb10, S2);
    S2 = wmma_f16(qa1, kb11, S2);

    // ---- online softmax update ----------------------------------------
    float corr[8];
#pragma unroll
    for (int j = 0; j < 8; ++j) {
      float cm = rmax16(fmaxf(S1[j], S2[j]));
      float mn = fmaxf(m[j], cm);
      corr[j] = __expf(m[j] - mn);
      float e1 = __expf(S1[j] - mn);
      float e2 = __expf(S2[j] - mn);
      s[j] = s[j] * corr[j] + rsum16(e1 + e2);
      m[j] = mn;
      O0[j] *= corr[j];
      O1[j] *= corr[j];
      const int r = j + 8 * half;
      Pbuf[wave][r * 32 + (lane & 15)]      = (_Float16)e1;
      Pbuf[wave][r * 32 + 16 + (lane & 15)] = (_Float16)e2;
    }

    // ---- O += P * V (P re-laid C-frag -> A-frag through LDS) ----------
    v16h pa  = load_a_f16(&Pbuf[wave][(lane & 15) * 32], 0, half);
    v16h vb0 = *(const v16h*)(vst + (lane & 15) * 32 + half * 16);
    v16h vb1 = *(const v16h*)(vst + (16 + (lane & 15)) * 32 + half * 16);
    O0 = wmma_f16(pa, vb0, O0);
    O1 = wmma_f16(pa, vb1, O1);
    nbuf ^= 1;
  }

  // ---- merge the 4 waves' partials ------------------------------------
#pragma unroll
  for (int j = 0; j < 8; ++j) {
    const int r = j + 8 * half;
    mbuf[wave][r] = m[j];
    sbuf[wave][r] = s[j];
    Obuf[wave][r][(lane & 15)]      = O0[j];
    Obuf[wave][r][16 + (lane & 15)] = O1[j];
  }
  __syncthreads();

  const int b = bh / HEADS, h = bh % HEADS;
  for (int idx = threadIdx.x; idx < 16 * 32; idx += 128) {
    const int r = idx >> 5, d = idx & 31;
    float M = mbuf[0][r];
#pragma unroll
    for (int w = 1; w < 4; ++w) M = fmaxf(M, mbuf[w][r]);
    float ssum = 0.0f, o = 0.0f;
#pragma unroll
    for (int w = 0; w < 4; ++w) {
      const float f = __expf(mbuf[w][r] - M);
      ssum += sbuf[w][r] * f;
      o += Obuf[w][r][d] * f;
    }
    omid[((size_t)b * NPAD + n0 + r) * CDIM + h * DHD + d] = o / ssum;
  }
}

// ---------------------------------------------------------------------------
// Kernel 4: output projection + residual. grid = B*19, 128 threads.
// ---------------------------------------------------------------------------
__global__ void __launch_bounds__(128)
outproj_kernel(const float* __restrict__ omid, const _Float16* __restrict__ Wo16,
               const float* __restrict__ bo, const float* __restrict__ query,
               float* __restrict__ out) {
  const int lane = threadIdx.x & 31, wave = threadIdx.x >> 5;
  const int half = lane >> 4;
  const int bb = blockIdx.x / NT;
  const int n0 = (blockIdx.x % NT) * 16;
  const float* rowp = omid + ((size_t)bb * NPAD + n0 + (lane & 15)) * CDIM;

  for (int ct = wave; ct < 16; ct += 4) {
    const int chan = ct * 16 + (lane & 15);
    v8f acc = gemm256(rowp, Wo16, chan, half);
    const float bias = bo[chan];
#pragma unroll
    for (int j = 0; j < 8; ++j) {
      const int n = n0 + j + 8 * half;
      if (n < NQ) {
        const size_t o = ((size_t)bb * NQ + n) * CDIM + chan;
        out[o] = acc[j] + bias + query[o];
      }
    }
  }
}

// ---------------------------------------------------------------------------
extern "C" void kernel_launch(void* const* d_in, const int* in_sizes, int n_in,
                              void* d_out, int out_size, void* d_ws, size_t ws_size,
                              hipStream_t stream) {
  const float* query      = (const float*)d_in[0];
  const float* key        = (const float*)d_in[1];
  const float* value      = (const float*)d_in[2];
  const float* query_pos  = (const float*)d_in[3];
  const float* key_pos    = (const float*)d_in[4];
  const float* query_sine = (const float*)d_in[5];
  const float* Wqc = (const float*)d_in[6];  const float* bqc = (const float*)d_in[7];
  const float* Wqp = (const float*)d_in[8];  const float* bqp = (const float*)d_in[9];
  const float* Wqs = (const float*)d_in[10]; const float* bqs = (const float*)d_in[11];
  const float* Wkc = (const float*)d_in[12]; const float* bkc = (const float*)d_in[13];
  const float* Wkp = (const float*)d_in[14]; const float* bkp = (const float*)d_in[15];
  const float* Wv  = (const float*)d_in[16]; const float* bv  = (const float*)d_in[17];
  const float* Wo  = (const float*)d_in[18]; const float* bo  = (const float*)d_in[19];
  float* out = (float*)d_out;

  auto aln = [](size_t x) { return (x + 255) & ~(size_t)255; };
  char* ws = (char*)d_ws;
  size_t off = 0;
  _Float16* W16  = (_Float16*)(ws + off); off += aln((size_t)7 * WROWS * sizeof(_Float16));
  _Float16* qcat = (_Float16*)(ws + off); off += aln((size_t)BATCH * HEADS * NPAD * 64 * sizeof(_Float16));
  _Float16* kcat = (_Float16*)(ws + off); off += aln((size_t)BATCH * HEADS * LKV  * 64 * sizeof(_Float16));
  _Float16* vT   = (_Float16*)(ws + off); off += aln((size_t)BATCH * HEADS * DHD  * LKV * sizeof(_Float16));
  float*    omid = (float*)   (ws + off); off += aln((size_t)BATCH * NPAD * CDIM * sizeof(float));
  (void)ws_size; (void)in_sizes; (void)n_in; (void)out_size;

  const _Float16* Wqc16 = W16 + 0 * WROWS;
  const _Float16* Wqp16 = W16 + 1 * WROWS;
  const _Float16* Wqs16 = W16 + 2 * WROWS;
  const _Float16* Wkc16 = W16 + 3 * WROWS;
  const _Float16* Wkp16 = W16 + 4 * WROWS;
  const _Float16* Wv16  = W16 + 5 * WROWS;
  const _Float16* Wo16  = W16 + 6 * WROWS;

  wconv_kernel<<<224, 256, 0, stream>>>(Wqc, Wqp, Wqs, Wkc, Wkp, Wv, Wo, W16);
  proj_kv_kernel<<<(BATCH * LKV) / 64, 128, 0, stream>>>(
      key, key_pos, value, Wkc16, Wkp16, Wv16, bkc, bkp, bv, kcat, vT);
  proj_q_kernel<<<BATCH * NT, 128, 0, stream>>>(
      query, query_pos, query_sine, Wqc16, Wqp16, Wqs16, bqc, bqp, bqs, qcat);
  attn_kernel<<<BATCH * HEADS * NT, 128, 0, stream>>>(qcat, kcat, vT, omid);
  outproj_kernel<<<BATCH * NT, 128, 0, stream>>>(omid, Wo16, bo, query, out);
}
